// ExplicitNeuralNetwork_56143812493596
// MI455X (gfx1250) — matrix operand order repaired and checked by reading, not compiled, not run on hardware
//
#include <hip/hip_runtime.h>
#include <hip/hip_bf16.h>
#include <math.h>

typedef __attribute__((ext_vector_type(16))) _Float16 v16h;
typedef __attribute__((ext_vector_type(8)))  _Float16 v8h;
typedef __attribute__((ext_vector_type(8)))  float    v8f;

#define NN 2048
#define II 64
#define OO 64
#define BB 64
#define TT 256
#define NBLK 128   // persistent grid: one 16-wide n-tile per block

__global__ __launch_bounds__(256)
void enn_init(float* __restrict__ st, float* __restrict__ nin32,
              _Float16* __restrict__ nin16, int* __restrict__ cnt) {
  int tid = blockIdx.x * blockDim.x + threadIdx.x;
  st[tid]    = 0.0f;
  nin32[tid] = 0.0f;
  nin16[tid] = (_Float16)0.0f;
  if (tid < TT) cnt[tid] = 0;      // grid-barrier arrive counters
}

// ---------------------------------------------------------------------------
// Persistent recurrent kernel. Block b owns n-tile n0 = b*16 and keeps its
// entire masked fp16 weight slice (16 cols x 2048 K = 64 KB) resident in LDS
// for all 256 timesteps: W is read from HBM exactly once, per-step B traffic
// is pure LDS. 8 wave32s: wave w -> m-tile (w&3), K-half (w>>2); halves are
// reduced through LDS, waves 0-3 run the epilogue. Steps are separated by a
// grid-wide atomic arrive barrier with device-scope fences.
//
// LDS B layout (two 16B planes per 32B fragment): entry
//   e = (kt>>1)*32 + 2*nl + (kt&1)
// so in the K-loop a lane (slot j = 2*nl+hi) reads plane offset
//   (kh*32+q)*512B + j*16B  -> 32 lanes cover one contiguous 512B span
// per ds_load_b128 = minimal 2-cycle LDS transaction, linear +512B stride.
// ---------------------------------------------------------------------------
__global__ __launch_bounds__(256)
void enn_persist(const float* __restrict__ W,
                 const unsigned char* __restrict__ mask,
                 const float* __restrict__ x,
                 const float* __restrict__ bias,
                 const float* __restrict__ alpha,
                 _Float16* __restrict__ nin16a, _Float16* __restrict__ nin16b,
                 float* __restrict__ nin32a, float* __restrict__ nin32b,
                 float* __restrict__ sta, float* __restrict__ stb,
                 int* __restrict__ cnt) {
  __shared__ _Float16 bpl0[128 * 16 * 8];            // 32 KB: K-halves 0..7
  __shared__ _Float16 bpl1[128 * 16 * 8];            // 32 KB: K-halves 8..15
  __shared__ float    red[4 * 32 * 8];               // 4 KB partial sums

  const int lane = threadIdx.x & 31;
  const int wave = threadIdx.x >> 5;                 // 0..7
  const int mi   = wave & 3;                         // m-tile index
  const int kh   = wave >> 2;                        // K half
  const int n0   = blockIdx.x * 16;                  // block-shared n-tile
  const int m0   = mi * 16;

  const int nl = lane & 15;
  const int hi = lane >> 4;
  const int mA = m0 + nl;                            // A row for this lane
  const int n  = n0 + nl;                            // output column
  const int j  = (nl << 1) | hi;                     // LDS B lane slot
  const bool isInput = (n0 < II);                    // uniform per block

  // ---- one-time: mask + convert this block's W slice into LDS ----
  if (!isInput) {
    for (int e = threadIdx.x; e < 2048; e += 256) {
      const int kt = ((e >> 5) << 1) | (e & 1);
      const int nn = (e >> 1) & 15;
      const float*         src = W    + (size_t)(n0 + nn) * NN + kt * 16;
      const unsigned char* msk = mask + (size_t)(n0 + nn) * NN + kt * 16;
      #pragma unroll
      for (int h = 0; h < 8; ++h) {
        bpl0[e * 8 + h] = (_Float16)(msk[h]     ? src[h]     : 0.0f);
        bpl1[e * 8 + h] = (_Float16)(msk[h + 8] ? src[h + 8] : 0.0f);
      }
    }
  }
  __syncthreads();

  const float bn = bias[n];
  const float an = alpha[n];

  for (int t = 0; t < TT; ++t) {
    const _Float16* ninOld16 = (t & 1) ? nin16b : nin16a;
    const float*    stOld    = (t & 1) ? stb    : sta;
    _Float16* ninNew16 = (t & 1) ? nin16a : nin16b;
    float*    ninNew32 = (t & 1) ? nin32a : nin32b;
    float*    stNew    = (t & 1) ? sta    : stb;

    v8f acc = {};
    if (!isInput) {
      // A (fp16 16x32): lane = row mA, K 0..15 contiguous (+16 for hi).
      const _Float16* aRow = ninOld16 + (size_t)mA * NN + kh * (NN / 2) + (hi ? 16 : 0);
      const _Float16* p0 = &bpl0[(kh * 32) * 256 + j * 8];
      const _Float16* p1 = &bpl1[(kh * 32) * 256 + j * 8];
      #pragma unroll 4
      for (int q = 0; q < 32; ++q) {                 // K = 32 per WMMA
        v8h alo = *(const v8h*)(aRow + q * 32);
        v8h ahi = *(const v8h*)(aRow + q * 32 + 8);
        v8h b0  = *(const v8h*)(p0 + q * 256);       // ds_load_b128
        v8h b1  = *(const v8h*)(p1 + q * 256);       // ds_load_b128
        v16h a, b;
        #pragma unroll
        for (int i = 0; i < 8; ++i) {
          a[i] = alo[i]; a[i + 8] = ahi[i];
          b[i] = b0[i];  b[i + 8] = b1[i];
        }
        acc = __builtin_amdgcn_wmma_f32_16x16x32_f16(
            false, a, false, b, (short)0, acc, false, false);
      }
      if (kh == 1) {                                 // upper K half -> LDS
        float* dst = &red[(mi * 32 + lane) * 8];
        #pragma unroll
        for (int r = 0; r < 8; ++r) dst[r] = acc[r];
      }
    }
    __syncthreads();

    if (kh == 0) {                                   // reduce + epilogue
      if (!isInput) {
        const float* p = &red[(mi * 32 + lane) * 8];
        #pragma unroll
        for (int r = 0; r < 8; ++r) acc[r] += p[r];
      }
      #pragma unroll
      for (int r = 0; r < 8; ++r) {
        const int mm = m0 + hi * 8 + r;              // C/D: vgpr r -> row
        const size_t idx = (size_t)mm * NN + n;
        float total;
        if (isInput) {
          total = x[((size_t)mm * TT + t) * II + n] + bn;      // x_t + bias
        } else {
          total = acc[r] + bn + tanhf(an * stOld[idx]);        // recurrent
        }
        const float ns  = total > 0.0f ? total : 0.01f * total; // leaky_relu
        const float nn2 = tanhf(total);
        stNew[idx]    = ns;
        ninNew32[idx] = nn2;
        ninNew16[idx] = (_Float16)nn2;               // next step's A matrix
      }
    }

    // ---- grid-wide step barrier (skip after the last step) ----
    if (t != TT - 1) {
      __threadfence();                               // release to device scope
      __syncthreads();
      if (threadIdx.x == 0) {
        __hip_atomic_fetch_add(&cnt[t], 1, __ATOMIC_ACQ_REL, __HIP_MEMORY_SCOPE_AGENT);
        while (__hip_atomic_load(&cnt[t], __ATOMIC_ACQUIRE, __HIP_MEMORY_SCOPE_AGENT)
               < (int)gridDim.x) {
          __builtin_amdgcn_s_sleep(1);
        }
      }
      __syncthreads();
      __threadfence();                               // acquire: drop stale lines
    }
  }
}

// ---------------------------------------------------------------------------
// d_out = [outputs (64x64) | states (64x2048) | nin (64x2048)], all fp32.
// ---------------------------------------------------------------------------
__global__ __launch_bounds__(256)
void enn_finalize(const float* __restrict__ nin32, const float* __restrict__ st32,
                  float* __restrict__ out) {
  int tid = blockIdx.x * blockDim.x + threadIdx.x;   // 0 .. 266239
  if (tid < BB * OO) {
    int b = tid >> 6, o = tid & 63;
    out[tid] = nin32[(size_t)b * NN + II + o];
  } else if (tid < BB * OO + BB * NN) {
    out[tid] = st32[tid - BB * OO];
  } else {
    out[tid] = nin32[tid - BB * OO - BB * NN];
  }
}

extern "C" void kernel_launch(void* const* d_in, const int* in_sizes, int n_in,
                              void* d_out, int out_size, void* d_ws, size_t ws_size,
                              hipStream_t stream) {
  const float*         x     = (const float*)d_in[0];         // (B,T,I)
  const float*         W     = (const float*)d_in[1];         // (N,N)
  const float*         bias  = (const float*)d_in[2];         // (N,)
  const float*         alpha = (const float*)d_in[3];         // (N,)
  const unsigned char* mask  = (const unsigned char*)d_in[4]; // (N,N) bool

  char* ws = (char*)d_ws;
  _Float16* nin16a = (_Float16*)(ws);                          // 256 KB
  _Float16* nin16b = (_Float16*)(ws +  262144);                // 256 KB
  float*    nin32a = (float*)   (ws +  524288);                // 512 KB
  float*    nin32b = (float*)   (ws + 1048576);                // 512 KB
  float*    sta    = (float*)   (ws + 1572864);                // 512 KB
  float*    stb    = (float*)   (ws + 2097152);                // 512 KB
  int*      cnt    = (int*)     (ws + 2621440);                // 256 ints

  enn_init<<<512, 256, 0, stream>>>(sta, nin32a, nin16a, cnt);

  enn_persist<<<NBLK, 256, 0, stream>>>(W, mask, x, bias, alpha,
                                        nin16a, nin16b, nin32a, nin32b,
                                        sta, stb, cnt);

  // After t=255 (odd) the final state lives in the "a" buffers.
  enn_finalize<<<1040, 256, 0, stream>>>(nin32a, sta, (float*)d_out);
}
